// MonotonicAttention_858993459466
// MI455X (gfx1250) — matrix operand order repaired and checked by reading, not compiled, not run on hardware
//
#include <hip/hip_runtime.h>
#include <stdint.h>

typedef _Float16 f16;
typedef _Float16 v16h __attribute__((ext_vector_type(16)));
typedef _Float16 v8h  __attribute__((ext_vector_type(8)));
typedef float    v8f  __attribute__((ext_vector_type(8)));

#define NB 32
#define NS 2048
#define ND 64
#define NWAVES 8
#define QW 16
#define QB (NWAVES * QW)   /* 128 query rows per workgroup */
#define KT 64              /* keys per iteration */
#define NITER (NS / KT)    /* 32 */
#define KP 72              /* LDS row pitch in halves: 144B = 16*9, conflict-free */

__device__ __forceinline__ v8h cvt8s(const float* __restrict__ p, float s) {
  const float4 a = *(const float4*)p;
  const float4 b = *(const float4*)(p + 4);
  v8h r;
  r[0] = (f16)(a.x * s); r[1] = (f16)(a.y * s); r[2] = (f16)(a.z * s); r[3] = (f16)(a.w * s);
  r[4] = (f16)(b.x * s); r[5] = (f16)(b.y * s); r[6] = (f16)(b.z * s); r[7] = (f16)(b.w * s);
  return r;
}

__device__ __forceinline__ v16h cat8(v8h a, v8h b) {
  v16h r;
#pragma unroll
  for (int i = 0; i < 8; ++i) { r[i] = a[i]; r[8 + i] = b[i]; }
  return r;
}

__device__ __forceinline__ uint32_t pk2(float a, float b) {
  union { f16 h[2]; uint32_t u; } z;
  z.h[0] = (f16)a; z.h[1] = (f16)b;
  return z.u;   // -> v_cvt_pk_f16_f32
}

#if defined(__has_builtin)
#if __has_builtin(__builtin_amdgcn_exp2f)
#define EXP2(x) __builtin_amdgcn_exp2f(x)
#else
#define EXP2(x) exp2f(x)
#endif
#else
#define EXP2(x) exp2f(x)
#endif

// two chained WMMAs: acc += A0 * B(rows[ofs..ofs+31]) + A1 * B(rows[32+ofs..])
__device__ __forceinline__ v8f mm2(const f16* row, int db, v16h a0, v16h a1, v8f acc) {
  const v16h b0 = cat8(*(const v8h*)(row + db),      *(const v8h*)(row + db + 16));
  const v16h b1 = cat8(*(const v8h*)(row + 32 + db), *(const v8h*)(row + 32 + db + 16));
  acc = __builtin_amdgcn_wmma_f32_16x16x32_f16(false, a0, false, b0, (short)0, acc, false, false);
  acc = __builtin_amdgcn_wmma_f32_16x16x32_f16(false, a1, false, b1, (short)0, acc, false, false);
  return acc;
}

__global__ __launch_bounds__(256)
void mono_attn_kernel(const float* __restrict__ Q, const float* __restrict__ K,
                      const float* __restrict__ V, const uint8_t* __restrict__ Mk,
                      const float* __restrict__ gamma, const int* __restrict__ dkp,
                      float* __restrict__ out)
{
  // 36 KB LDS, all rows padded to 144B so stride-1-row lane access is conflict-free
  __shared__ __align__(32) f16 Ks[KT][KP];          // permuted-key rows x depth (f16, pre-scaled)
  __shared__ __align__(32) f16 Vt[ND][KP];          // depth x key (natural key order)
  __shared__ __align__(32) f16 Pb[NWAVES][16][KP];  // per-wave P, natural key order

  const int tid  = threadIdx.x;
  const int wave = tid >> 5;
  const int lane = tid & 31;
  const int hf   = lane >> 4;
  const int ln   = lane & 15;

  const int bIdx = blockIdx.x / (NS / QB);
  const int q0wg = (blockIdx.x % (NS / QB)) * QB;
  const int q0   = q0wg + wave * QW;

  // d = -softplus(gamma); work in log2 domain: scl2 = d*log2(e)/sqrt(dk) <= 0
  const float g    = gamma[0];
  const float sp   = fmaxf(g, 0.0f) + log1pf(__expf(-fabsf(g)));
  const float scl2 = -sp * rsqrtf((float)dkp[0]) * 1.4426950408889634f;
  const float sm2  = scl2 * (-1.0e8f);   // masked score (log2 domain)
  const float ka   = sqrtf(-scl2);       // fold scale into data: qa*ka = scl2
  const float qa   = -ka;

  // ---- Q A-fragments (pre-scaled): halves 0-7 = K[c0..c0+7], 8-15 = K[c0+16..c0+23]
  v16h aq0, aq1;
  {
    const float* qrow = Q + ((size_t)bIdx * NS + (size_t)(q0 + ln)) * ND;
    const int c0 = hf * 8;
    aq0 = cat8(cvt8s(qrow + c0, qa),      cvt8s(qrow + c0 + 16, qa));
    aq1 = cat8(cvt8s(qrow + 32 + c0, qa), cvt8s(qrow + 32 + c0 + 16, qa));
  }

  float mrow[8], lrow[8];   // lrow = per-lane partial sum (reduced once at end)
  v8f o[4];
#pragma unroll
  for (int r = 0; r < 8; ++r) { mrow[r] = -INFINITY; lrow[r] = 0.0f; }
#pragma unroll
  for (int c = 0; c < 4; ++c)
#pragma unroll
    for (int r = 0; r < 8; ++r) o[c][r] = 0.0f;

  // staging assignments
  const int krS = tid >> 2;                              // K: key row, 16 cols/thread
  const int kcS = (tid & 3) * 16;
  const int krP = ((krS & 3) << 4) | (krS >> 2);         // permuted row: tile t col n <-> key 4n+t
  const int vcS = tid & 63;                              // V: one depth-column, 16 keys/thread
  const int vqS = (tid >> 6) * 16;

  // mask: lane covers keys 4*ln..4*ln+3 of its rows -> one coalesced b32 per row
  const uint8_t* mb = Mk + ((size_t)bIdx * NS + (size_t)(q0 + hf * 8)) * NS + 4 * ln;

#pragma unroll 1
  for (int it = 0; it < NITER; ++it) {
    const int k0 = it * KT;

    uint32_t mw[8];
#pragma unroll
    for (int r = 0; r < 8; ++r)
      mw[r] = __builtin_nontemporal_load((const uint32_t*)(mb + (size_t)r * NS + k0));

    __syncthreads();   // previous tile fully consumed
    {
      const float* kp = K + ((size_t)bIdx * NS + (size_t)(k0 + krS)) * ND + kcS;
      *(v8h*)&Ks[krP][kcS]     = cvt8s(kp, ka);
      *(v8h*)&Ks[krP][kcS + 8] = cvt8s(kp + 8, ka);

      const float* vp = V + ((size_t)bIdx * NS + (size_t)(k0 + vqS)) * ND + vcS;
      v8h t0, t1;
#pragma unroll
      for (int i = 0; i < 8; ++i) t0[i] = (f16)vp[(size_t)i * ND];
#pragma unroll
      for (int i = 0; i < 8; ++i) t1[i] = (f16)vp[(size_t)(8 + i) * ND];
      *(v8h*)&Vt[vcS][vqS]     = t0;
      *(v8h*)&Vt[vcS][vqS + 8] = t1;

      if (it + 1 < NITER) {
        __builtin_prefetch(kp + KT * ND, 0, 3);
        __builtin_prefetch(vp + KT * ND, 0, 3);
      }
    }
    __syncthreads();

    // ---- scores: 4 tiles; tile t col ln = key 4*ln+t (LDS row 16t+ln) ----
    v8f s0, s1, s2, s3;
#pragma unroll
    for (int i = 0; i < 8; ++i) { s0[i] = 0; s1[i] = 0; s2[i] = 0; s3[i] = 0; }
    const int db = hf * 8;
    s0 = mm2(&Ks[ln][0],      db, aq0, aq1, s0);
    s1 = mm2(&Ks[16 + ln][0], db, aq0, aq1, s1);
    s2 = mm2(&Ks[32 + ln][0], db, aq0, aq1, s2);
    s3 = mm2(&Ks[48 + ln][0], db, aq0, aq1, s3);

    // ---- mask + online softmax ----
    float alpha[8];
#pragma unroll
    for (int r = 0; r < 8; ++r) {
      const uint32_t w = mw[r];
      const float e0 = (w & 0x000000FFu) ? sm2 : s0[r];
      const float e1 = (w & 0x0000FF00u) ? sm2 : s1[r];
      const float e2 = (w & 0x00FF0000u) ? sm2 : s2[r];
      const float e3 = (w & 0xFF000000u) ? sm2 : s3[r];
      float mx = fmaxf(fmaxf(e0, e1), fmaxf(e2, e3));
      mx = fmaxf(mx, __shfl_xor(mx, 1, 32));
      mx = fmaxf(mx, __shfl_xor(mx, 2, 32));
      mx = fmaxf(mx, __shfl_xor(mx, 4, 32));
      mx = fmaxf(mx, __shfl_xor(mx, 8, 32));
      const float mnew = fmaxf(mrow[r], mx);
      const float a  = EXP2(mrow[r] - mnew);
      const float p0 = EXP2(e0 - mnew);
      const float p1 = EXP2(e1 - mnew);
      const float p2 = EXP2(e2 - mnew);
      const float p3 = EXP2(e3 - mnew);
      lrow[r]  = lrow[r] * a + ((p0 + p1) + (p2 + p3));
      mrow[r]  = mnew;
      alpha[r] = a;
      uint2 pw; pw.x = pk2(p0, p1); pw.y = pk2(p2, p3);
      *(uint2*)&Pb[wave][r + hf * 8][4 * ln] = pw;   // one ds_store_b64, natural key order
    }

    // ---- O = alpha*O + P(16x64) . V_tile(64x64) ----
    const int c0 = hf * 8;
    const v16h ap0 = cat8(*(const v8h*)&Pb[wave][ln][c0],      *(const v8h*)&Pb[wave][ln][c0 + 16]);
    const v16h ap1 = cat8(*(const v8h*)&Pb[wave][ln][32 + c0], *(const v8h*)&Pb[wave][ln][32 + c0 + 16]);
#pragma unroll
    for (int c = 0; c < 4; ++c) {
      v8f oc = o[c];
#pragma unroll
      for (int r = 0; r < 8; ++r) oc[r] *= alpha[r];
      o[c] = mm2(&Vt[c * 16 + ln][0], db, ap0, ap1, oc);
    }
  }

  // ---- epilogue: reduce row sums once, divide, NT store ----
#pragma unroll
  for (int r = 0; r < 8; ++r) {
    float l = lrow[r];
    l += __shfl_xor(l, 1, 32);
    l += __shfl_xor(l, 2, 32);
    l += __shfl_xor(l, 4, 32);
    l += __shfl_xor(l, 8, 32);
    const float inv = 1.0f / l;
    float* op = out + ((size_t)bIdx * NS + (size_t)(q0 + r + hf * 8)) * ND + ln;
    __builtin_nontemporal_store(o[0][r] * inv, op);
    __builtin_nontemporal_store(o[1][r] * inv, op + 16);
    __builtin_nontemporal_store(o[2][r] * inv, op + 32);
    __builtin_nontemporal_store(o[3][r] * inv, op + 48);
  }
}

extern "C" void kernel_launch(void* const* d_in, const int* in_sizes, int n_in,
                              void* d_out, int out_size, void* d_ws, size_t ws_size,
                              hipStream_t stream) {
  (void)in_sizes; (void)n_in; (void)out_size; (void)d_ws; (void)ws_size;
  const float*   Q  = (const float*)d_in[0];
  const float*   K  = (const float*)d_in[1];
  const float*   V  = (const float*)d_in[2];
  const uint8_t* Mk = (const uint8_t*)d_in[3];   // jax bool -> 1 byte/elem
  const float*   g  = (const float*)d_in[4];
  const int*     dk = (const int*)d_in[5];
  float* out = (float*)d_out;

  dim3 grid(NB * (NS / QB));   // 512 workgroups
  dim3 block(256);             // 8 wave32
  hipLaunchKernelGGL(mono_attn_kernel, grid, block, 0, stream,
                     Q, K, V, Mk, g, dk, out);
}
